// TimeCrystalSequenceBlock_18356690223128
// MI455X (gfx1250) — compile-verified, matrix operand order checked by reading, not run on hardware
//
#include <hip/hip_runtime.h>
#include <math.h>

// ---------------------------------------------------------------------------
// Types for CDNA5 WMMA (wave32, 16x16x32 bf16 -> f32) and TDM descriptors
// ---------------------------------------------------------------------------
typedef __bf16 bf16_t;
typedef __attribute__((ext_vector_type(8)))  bf16_t v8bf;
typedef __attribute__((ext_vector_type(16))) bf16_t v16bf;
typedef __attribute__((ext_vector_type(8)))  float  v8f;
typedef __attribute__((ext_vector_type(4)))  unsigned int u32x4;
typedef __attribute__((ext_vector_type(8)))  int i32x8;
typedef __attribute__((ext_vector_type(4)))  int i32x4;

#if __has_builtin(__builtin_amdgcn_tensor_load_to_lds) && \
    __has_builtin(__builtin_amdgcn_s_wait_tensorcnt)
#define USE_TDM 1
#else
#define USE_TDM 0
#endif

__device__ __forceinline__ float  bf2f(bf16_t v) { return (float)v; }
__device__ __forceinline__ bf16_t f2bf(float v)  { return (bf16_t)v; }

__device__ __forceinline__ float softplusf(float x) {
  return x > 20.0f ? x : log1pf(expf(x));
}

// ---------------------------------------------------------------------------
// Problem constants
// ---------------------------------------------------------------------------
#define SDIM    512
#define DH      2048
#define IDIM    1024
#define ZDIM    2048          // q(512) + p(512) + x(1024)
#define NQ      6
#define QDIM    64            // 2^6

// GEMM tiling
#define BM   128
#define BN   128
#define BKK  32
#define LDS_S 40              // 32 + 8 pad (bf16); 80B row stride (TDM pad 64B+16B)

// Epilogue selectors
#define EPI_TANH      0       // Cb = bf16(tanh(acc + bias[n]))
#define EPI_TANH_DW3  1       // Cb = bf16((1 - tanh(acc+bias[n])^2) * w3[n])
#define EPI_OMH2      2       // Cb = bf16((1 - Hb[m,n]^2) * acc)
#define EPI_AXPY      3       // Cf = base + alpha*dt*acc ; Cb (+Cb2) = bf16(Cf)
#define EPI_BIASF     4       // Cf = acc + bias[n]

#if USE_TDM
// ---------------------------------------------------------------------------
// TDM: issue a 128-row x 32-elem bf16 tile load, row stride ld (elements),
// into LDS at lds_byte_addr with 80-byte padded rows (pad 16B every 64B).
// D# packing per CDNA5 ISA.
// Toolchain here exposes the 6-arg builtin:
//   (u32x4 g0, i32x8 g1, i32x4 g2, i32x4 g3, i32x8 g4, i32 cpol)
// ---------------------------------------------------------------------------
__device__ __forceinline__ void tdm_load_tile(const bf16_t* gptr, int ld,
                                              unsigned lds_byte_addr) {
  const unsigned long long ga = (unsigned long long)(uintptr_t)gptr;
  u32x4 g0;
  g0.x = 1u;                                    // count=1, user mode, no gather
  g0.y = lds_byte_addr;                         // lds_addr
  g0.z = (unsigned)(ga & 0xFFFFFFFFu);          // global_addr[31:0]
  g0.w = (unsigned)((ga >> 32) & 0x1FFFFFFu)    // global_addr[56:32]
       | (2u << 30);                            // type=2 ("image")
  const unsigned td0 = (unsigned)ld;            // tensor_dim0 (row length)
  const unsigned td1 = 1u << 20;                // tensor_dim1 (ample; never OOB)
  i32x8 g1;
  g1[0] = (int)((1u << 16)                      // data_size = 2 bytes
              | (1u << 20)                      // pad_enable
              | (3u << 22)                      // pad_interval: 16 DWORDs (64B)
              | (3u << 25));                    // pad_amount: 4 DWORDs (16B)
  g1[1] = (int)((td0 & 0xFFFFu) << 16);         // [31:16]=tensor_dim0 lo16
  g1[2] = (int)(((td0 >> 16) & 0xFFFFu) | ((td1 & 0xFFFFu) << 16));
  g1[3] = (int)(((td1 >> 16) & 0xFFFFu) | (32u << 16));   // tile_dim0 = 32
  g1[4] = (int)(128u);                          // tile_dim1 = 128, tile_dim2 = 0
  g1[5] = (int)((unsigned)ld);                  // tensor_dim0_stride[31:0]
  g1[6] = 0;                                    // stride[47:32]=0, dim1_stride lo=0
  g1[7] = 0;
  const i32x4 gz4 = {0, 0, 0, 0};               // 2D tensor: groups 2/3 unused
  const i32x8 gz8 = {0, 0, 0, 0, 0, 0, 0, 0};
  __builtin_amdgcn_tensor_load_to_lds(g0, g1, gz4, gz4, gz8, 0);
}
#endif

// ---------------------------------------------------------------------------
// WMMA GEMM:  C[M,N] = A[M,K] * Bt[N,K]^T   (both operands K-contiguous bf16)
// Block = 256 threads (8 wave32), 128x128 tile, BK=32.
// Waves 4x2; each wave: 2 m-tiles x 4 n-tiles of 16x16 => 8 v_wmma per k-step.
// TDM path: wave 0 drives the Tensor Data Mover into double-buffered LDS.
// ---------------------------------------------------------------------------
template<int EPI>
__global__ __launch_bounds__(256)
void wmma_gemm(const bf16_t* __restrict__ A,  int lda,
               const bf16_t* __restrict__ Bt, int ldb,
               int K,
               float*  __restrict__ Cf,  int ldcf,
               bf16_t* __restrict__ Cb,  int ldcb,
               bf16_t* __restrict__ Cb2, int ldcb2,
               const float* __restrict__ bias,
               const float* __restrict__ w3,
               const bf16_t* __restrict__ Hb, int ldh,
               const float* __restrict__ basef, int ldbase,
               const float* __restrict__ dtp, float alpha)
{
  const int tid   = threadIdx.x;
  const int lane  = tid & 31;
  const int wave  = tid >> 5;
  const int wm    = wave >> 1;        // 0..3 -> 32-row strip
  const int wn    = wave & 1;         // 0..1 -> 64-col strip
  const int lcol  = lane & 15;
  const int lhalf = lane >> 4;        // 0 or 1
  const int bm    = blockIdx.y * BM;
  const int bn    = blockIdx.x * BN;

  v8f acc[2][4] = {};

#if USE_TDM
  __shared__ bf16_t As[2][BM * LDS_S];
  __shared__ bf16_t Bs[2][BN * LDS_S];

  const bf16_t* Ab = A  + (size_t)bm * lda;
  const bf16_t* Bb = Bt + (size_t)bn * ldb;

  // Prologue: TDM-load tile 0 into buffer 0 (wave 0 only; TENSORcnt-tracked).
  if (wave == 0) {
    tdm_load_tile(Ab, lda, (unsigned)(uintptr_t)&As[0][0]);
    tdm_load_tile(Bb, ldb, (unsigned)(uintptr_t)&Bs[0][0]);
    __builtin_amdgcn_s_wait_tensorcnt(0);
  }
  __syncthreads();

  int cur = 0;
  for (int kt = 0; kt < K; kt += BKK) {
    // Kick off next tile into the other buffer while this one is consumed.
    if (wave == 0 && (kt + BKK) < K) {
      tdm_load_tile(Ab + kt + BKK, lda, (unsigned)(uintptr_t)&As[cur ^ 1][0]);
      tdm_load_tile(Bb + kt + BKK, ldb, (unsigned)(uintptr_t)&Bs[cur ^ 1][0]);
    }

    // Fragments (ISA 16-bit A layout: lane<16 K0-7/16-23, lane>=16 K8-15/24-31)
    v16bf af[2], bfg[4];
    #pragma unroll
    for (int im = 0; im < 2; ++im) {
      const int ar = wm * 32 + im * 16 + lcol;
      const v8bf lo = *(const v8bf*)&As[cur][ar * LDS_S + lhalf * 8];
      const v8bf hi = *(const v8bf*)&As[cur][ar * LDS_S + lhalf * 8 + 16];
      af[im] = __builtin_shufflevector(lo, hi, 0,1,2,3,4,5,6,7,8,9,10,11,12,13,14,15);
    }
    #pragma unroll
    for (int jn = 0; jn < 4; ++jn) {
      const int br = wn * 64 + jn * 16 + lcol;
      const v8bf lo = *(const v8bf*)&Bs[cur][br * LDS_S + lhalf * 8];
      const v8bf hi = *(const v8bf*)&Bs[cur][br * LDS_S + lhalf * 8 + 16];
      bfg[jn] = __builtin_shufflevector(lo, hi, 0,1,2,3,4,5,6,7,8,9,10,11,12,13,14,15);
    }

    #pragma unroll
    for (int im = 0; im < 2; ++im)
      #pragma unroll
      for (int jn = 0; jn < 4; ++jn)
        acc[im][jn] = __builtin_amdgcn_wmma_f32_16x16x32_bf16(
            false, af[im], false, bfg[jn], (short)0, acc[im][jn], false, false);

    // Publish next buffer / protect current one, then flip.
    if (wave == 0) __builtin_amdgcn_s_wait_tensorcnt(0);
    __syncthreads();
    cur ^= 1;
  }
#else
  __shared__ bf16_t As[BM * LDS_S];
  __shared__ bf16_t Bs[BN * LDS_S];

  const int lrow = tid >> 2;          // 0..63
  const int lc8  = (tid & 3) * 8;     // 0,8,16,24

  for (int kt = 0; kt < K; kt += BKK) {
    #pragma unroll
    for (int i = 0; i < 2; ++i) {
      const int row = lrow + 64 * i;
      const v8bf av = *(const v8bf*)(A  + (size_t)(bm + row) * lda + kt + lc8);
      const v8bf bv = *(const v8bf*)(Bt + (size_t)(bn + row) * ldb + kt + lc8);
      *(v8bf*)&As[row * LDS_S + lc8] = av;
      *(v8bf*)&Bs[row * LDS_S + lc8] = bv;
    }
    __syncthreads();

    v16bf af[2], bfg[4];
    #pragma unroll
    for (int im = 0; im < 2; ++im) {
      const int ar = wm * 32 + im * 16 + lcol;
      const v8bf lo = *(const v8bf*)&As[ar * LDS_S + lhalf * 8];
      const v8bf hi = *(const v8bf*)&As[ar * LDS_S + lhalf * 8 + 16];
      af[im] = __builtin_shufflevector(lo, hi, 0,1,2,3,4,5,6,7,8,9,10,11,12,13,14,15);
    }
    #pragma unroll
    for (int jn = 0; jn < 4; ++jn) {
      const int br = wn * 64 + jn * 16 + lcol;
      const v8bf lo = *(const v8bf*)&Bs[br * LDS_S + lhalf * 8];
      const v8bf hi = *(const v8bf*)&Bs[br * LDS_S + lhalf * 8 + 16];
      bfg[jn] = __builtin_shufflevector(lo, hi, 0,1,2,3,4,5,6,7,8,9,10,11,12,13,14,15);
    }

    #pragma unroll
    for (int im = 0; im < 2; ++im)
      #pragma unroll
      for (int jn = 0; jn < 4; ++jn)
        acc[im][jn] = __builtin_amdgcn_wmma_f32_16x16x32_bf16(
            false, af[im], false, bfg[jn], (short)0, acc[im][jn], false, false);

    __syncthreads();
  }
#endif

  const float dtv = (EPI == EPI_AXPY) ? dtp[0] : 0.0f;

  // ---- epilogue (C layout: VGPR r -> M = r + 8*lhalf, lane&15 -> N) ----
  #pragma unroll
  for (int im = 0; im < 2; ++im) {
    #pragma unroll
    for (int jn = 0; jn < 4; ++jn) {
      const int n  = bn + wn * 64 + jn * 16 + lcol;
      const int mb = bm + wm * 32 + im * 16 + lhalf * 8;
      #pragma unroll
      for (int r = 0; r < 8; ++r) {
        const int m = mb + r;
        const float v = acc[im][jn][r];
        if constexpr (EPI == EPI_TANH) {
          const float h = tanhf(v + bias[n]);
          Cb[(size_t)m * ldcb + n] = f2bf(h);
        } else if constexpr (EPI == EPI_TANH_DW3) {
          const float h = tanhf(v + bias[n]);
          Cb[(size_t)m * ldcb + n] = f2bf((1.0f - h * h) * w3[n]);
        } else if constexpr (EPI == EPI_OMH2) {
          const float h = bf2f(Hb[(size_t)m * ldh + n]);
          Cb[(size_t)m * ldcb + n] = f2bf((1.0f - h * h) * v);
        } else if constexpr (EPI == EPI_AXPY) {
          const float res = basef[(size_t)m * ldbase + n] + alpha * dtv * v;
          Cf[(size_t)m * ldcf + n] = res;
          Cb[(size_t)m * ldcb + n] = f2bf(res);
          if (Cb2) Cb2[(size_t)m * ldcb2 + n] = f2bf(res);
        } else { // EPI_BIASF
          Cf[(size_t)m * ldcf + n] = v + bias[n];
        }
      }
    }
  }
}

// ---------------------------------------------------------------------------
// Weight prep: f32 -> bf16 convert, and transpose-convert
// ---------------------------------------------------------------------------
__global__ void k_conv(const float* __restrict__ src, bf16_t* __restrict__ dst, int n) {
  for (int i = blockIdx.x * blockDim.x + threadIdx.x; i < n;
       i += gridDim.x * blockDim.x)
    dst[i] = f2bf(src[i]);
}

__global__ __launch_bounds__(256)
void k_transconv(const float* __restrict__ src, bf16_t* __restrict__ dst,
                 int R, int C) {  // src RxC f32 -> dst CxR bf16
  __shared__ float t[32][33];
  const int bx = blockIdx.x * 32, by = blockIdx.y * 32;
  const int tx = threadIdx.x, ty = threadIdx.y;  // block (32,8)
  #pragma unroll
  for (int i = 0; i < 32; i += 8)
    t[ty + i][tx] = src[(size_t)(by + ty + i) * C + bx + tx];
  __syncthreads();
  #pragma unroll
  for (int i = 0; i < 32; i += 8)
    dst[(size_t)(bx + ty + i) * R + by + tx] = f2bf(t[tx][ty + i]);
}

// ---------------------------------------------------------------------------
// Unpack h_padded -> q,p (f32) and z = [q,p,x] (bf16)
// ---------------------------------------------------------------------------
__global__ void k_build(const float* __restrict__ x, const float* __restrict__ hp,
                        float* __restrict__ q, float* __restrict__ p,
                        bf16_t* __restrict__ z, int B) {
  const int idx = blockIdx.x * blockDim.x + threadIdx.x;
  if (idx >= B * ZDIM) return;
  const int b = idx >> 11, j = idx & 2047;
  float v;
  if (j < 1024) {
    v = hp[(size_t)b * 2048 + j];     // first 8 of 16 slices = 1024 contiguous
    if (j < 512) q[(size_t)b * SDIM + j] = v;
    else         p[(size_t)b * SDIM + (j - 512)] = v;
  } else {
    v = x[(size_t)b * IDIM + (j - 1024)];
  }
  z[idx] = f2bf(v);
}

// ---------------------------------------------------------------------------
// VQC: 6-qubit real-amplitude simulator, one sample per thread, state in VGPRs
// ---------------------------------------------------------------------------
template<int KQ>
__device__ __forceinline__ void apply_ry(float* st, float a) {
  const float cv = cosf(0.5f * a), sv = sinf(0.5f * a);
  constexpr int stride = 1 << (5 - KQ);
  #pragma unroll
  for (int i = 0; i < QDIM; ++i) {
    if ((i & stride) == 0) {
      const float a0 = st[i], a1 = st[i + stride];
      st[i]          = cv * a0 - sv * a1;
      st[i + stride] = sv * a0 + cv * a1;
    }
  }
}

__device__ __forceinline__ void ry_layer(float* st, const float* a) {
  apply_ry<0>(st, a[0]); apply_ry<1>(st, a[1]); apply_ry<2>(st, a[2]);
  apply_ry<3>(st, a[3]); apply_ry<4>(st, a[4]); apply_ry<5>(st, a[5]);
}

__global__ __launch_bounds__(256)
void k_vqc(const float* __restrict__ x, const float* __restrict__ q,
           const float* __restrict__ p, const float* __restrict__ encW,
           const float* __restrict__ encb, const float* __restrict__ thetas,
           const float* __restrict__ gain, const float* __restrict__ shift,
           float* __restrict__ acc, int B) {
  const int b = blockIdx.x * blockDim.x + threadIdx.x;
  if (b >= B) return;

  float ang[NQ];
  #pragma unroll
  for (int j = 0; j < NQ; ++j) ang[j] = encb[j];

  const float* xr = x + (size_t)b * IDIM;
  const float* qr = q + (size_t)b * SDIM;
  const float* pr = p + (size_t)b * SDIM;
  for (int i = 0; i < IDIM; ++i) {         // ctrl = [x, q, p]
    const float c = xr[i];
    #pragma unroll
    for (int j = 0; j < NQ; ++j) ang[j] += c * encW[i * NQ + j];
  }
  for (int i = 0; i < SDIM; ++i) {
    const float c = qr[i];
    #pragma unroll
    for (int j = 0; j < NQ; ++j) ang[j] += c * encW[(IDIM + i) * NQ + j];
  }
  for (int i = 0; i < SDIM; ++i) {
    const float c = pr[i];
    #pragma unroll
    for (int j = 0; j < NQ; ++j) ang[j] += c * encW[(IDIM + SDIM + i) * NQ + j];
  }

  float st[QDIM];
  #pragma unroll
  for (int i = 0; i < QDIM; ++i) st[i] = 0.0f;
  st[0] = 1.0f;

  ry_layer(st, ang);                        // data encoding

  #pragma unroll
  for (int l = 0; l < 3; ++l) {             // variational layers
    float th[NQ];
    #pragma unroll
    for (int k = 0; k < NQ; ++k) th[k] = thetas[l * NQ + k];
    ry_layer(st, th);
    #pragma unroll
    for (int i = 0; i < QDIM; ++i) {        // ring-CZ sign
      int cnt = 0;
      #pragma unroll
      for (int k = 0; k < NQ; ++k) {
        const int b0 = (i >> (5 - k)) & 1;
        const int b1 = (i >> (5 - ((k + 1) % NQ))) & 1;
        cnt += b0 & b1;
      }
      if (cnt & 1) st[i] = -st[i];
    }
  }

  float sample = 0.0f;
  #pragma unroll
  for (int i = 0; i < QDIM; ++i) {
    const float zo = (6.0f - 2.0f * (float)__popc(i)) * (1.0f / 6.0f);
    sample += st[i] * st[i] * zo;
  }

  const float sg = softplusf(gain[0]);
  const float ss = softplusf(shift[0]);
  atomicAdd(acc, softplusf(sg * sample - ss));
}

__global__ void k_zero(float* acc) { acc[0] = 0.0f; }

__global__ void k_dt(const float* __restrict__ acc, const float* __restrict__ time_bias,
                     const float* __restrict__ cap_param,
                     const float* __restrict__ last_stable,
                     float* __restrict__ dt, int B) {
  const float raw = acc[0] / (float)B + time_bias[0];
  const float cap = fmaxf(softplusf(cap_param[0]), 1e-6f);
  float evo = cap / (1.0f + expf(-raw / cap));
  if (!(evo - evo == 0.0f)) evo = cap / (1.0f + expf(-time_bias[0] / cap));
  const float prev = fmaxf(last_stable[0], 1e-6f);
  float cand = fmaxf(evo, 1e-6f);
  cand = fminf(cand, prev + fmaxf(prev * 0.1f, 1e-4f));
  cand = fminf(cand, 0.05f);
  dt[0] = (cand - cand == 0.0f) ? cand : prev;
}

// ---------------------------------------------------------------------------
// Pack h_next: slices 0..7 = [q_next, p_next], slices 8..15 = 0
// ---------------------------------------------------------------------------
__global__ void k_pack(const float* __restrict__ qn, const float* __restrict__ pn,
                       float* __restrict__ hout, int B) {
  const int idx = blockIdx.x * blockDim.x + threadIdx.x;
  if (idx >= B * 2048) return;
  const int b = idx >> 11, j = idx & 2047;
  float v = 0.0f;
  if (j < 512)       v = qn[(size_t)b * SDIM + j];
  else if (j < 1024) v = pn[(size_t)b * SDIM + (j - 512)];
  hout[idx] = v;
}

// ---------------------------------------------------------------------------
// Host orchestration
// ---------------------------------------------------------------------------
extern "C" void kernel_launch(void* const* d_in, const int* in_sizes, int n_in,
                              void* d_out, int out_size, void* d_ws, size_t ws_size,
                              hipStream_t stream) {
  const float* x       = (const float*)d_in[0];
  const float* hp      = (const float*)d_in[1];
  const float* W1      = (const float*)d_in[3];
  const float* b1      = (const float*)d_in[4];
  const float* W2      = (const float*)d_in[5];
  const float* b2      = (const float*)d_in[6];
  const float* W3      = (const float*)d_in[7];
  const float* W_out   = (const float*)d_in[9];
  const float* b_out   = (const float*)d_in[10];
  const float* encW    = (const float*)d_in[11];
  const float* encb    = (const float*)d_in[12];
  const float* thetas  = (const float*)d_in[13];
  const float* gain    = (const float*)d_in[14];
  const float* shift   = (const float*)d_in[15];
  const float* cap     = (const float*)d_in[16];
  const float* tbias   = (const float*)d_in[17];
  const float* lstable = (const float*)d_in[18];

  const int B = in_sizes[0] / IDIM;   // 2048

  // ---- carve workspace ----
  char* wsp = (char*)d_ws;
  auto alloc = [&](size_t bytes) -> void* {
    void* ptr = (void*)wsp;
    wsp += (bytes + 255) & ~(size_t)255;
    return ptr;
  };
  float*  s_acc  = (float*)alloc(sizeof(float));
  float*  s_dt   = (float*)alloc(sizeof(float));
  float*  qf     = (float*)alloc((size_t)B * SDIM * 4);
  float*  pf     = (float*)alloc((size_t)B * SDIM * 4);
  float*  p_half = (float*)alloc((size_t)B * SDIM * 4);
  float*  q_next = (float*)alloc((size_t)B * SDIM * 4);
  float*  p_next = (float*)alloc((size_t)B * SDIM * 4);
  bf16_t* W1bf   = (bf16_t*)alloc((size_t)DH * DH * 2);
  bf16_t* W2bf   = (bf16_t*)alloc((size_t)DH * DH * 2);
  bf16_t* W1T    = (bf16_t*)alloc((size_t)DH * DH * 2);
  bf16_t* W2T    = (bf16_t*)alloc((size_t)DH * DH * 2);
  bf16_t* WoT    = (bf16_t*)alloc((size_t)IDIM * IDIM * 2);
  bf16_t* zbf    = (bf16_t*)alloc((size_t)B * ZDIM * 2);
  bf16_t* h1bf   = (bf16_t*)alloc((size_t)B * DH * 2);
  bf16_t* g2bf   = (bf16_t*)alloc((size_t)B * DH * 2);
  bf16_t* g1bf   = (bf16_t*)alloc((size_t)B * DH * 2);
  bf16_t* qpbf   = (bf16_t*)alloc((size_t)B * IDIM * 2);

  float* out_main = (float*)d_out;                       // (B, 1024)
  float* out_hnxt = (float*)d_out + (size_t)B * IDIM;    // (B, 16, 128)

  // ---- prep: weights -> bf16 (+ transposes) ----
  k_zero<<<1, 1, 0, stream>>>(s_acc);
  k_conv<<<2048, 256, 0, stream>>>(W1, W1bf, DH * DH);
  k_conv<<<2048, 256, 0, stream>>>(W2, W2bf, DH * DH);
  k_transconv<<<dim3(DH / 32, DH / 32), dim3(32, 8), 0, stream>>>(W1, W1T, DH, DH);
  k_transconv<<<dim3(DH / 32, DH / 32), dim3(32, 8), 0, stream>>>(W2, W2T, DH, DH);
  k_transconv<<<dim3(IDIM / 32, IDIM / 32), dim3(32, 8), 0, stream>>>(W_out, WoT, IDIM, IDIM);

  // ---- unpack state, build z, compute dt ----
  const int nBZ = (B * ZDIM + 255) / 256;
  k_build<<<nBZ, 256, 0, stream>>>(x, hp, qf, pf, zbf, B);
  k_vqc<<<B / 256, 256, 0, stream>>>(x, qf, pf, encW, encb, thetas, gain, shift, s_acc, B);
  k_dt<<<1, 1, 0, stream>>>(s_acc, tbias, cap, lstable, s_dt, B);

  // ---- leapfrog: three gradient evaluations ----
  const dim3 gFull(DH / BN, B / BM);     // 2048-wide GEMMs
  const dim3 gHalf(SDIM / BN, B / BM);   // 512-wide gradient slice
  const dim3 gOut(IDIM / BN, B / BM);

  for (int ev = 0; ev < 3; ++ev) {
    // h1 = tanh(z @ W1 + b1)
    wmma_gemm<EPI_TANH><<<gFull, 256, 0, stream>>>(
        zbf, ZDIM, W1T, DH, ZDIM, nullptr, 0, h1bf, DH, nullptr, 0,
        b1, nullptr, nullptr, 0, nullptr, 0, nullptr, 0.0f);
    // g2 = (1 - tanh(h1 @ W2 + b2)^2) * W3[:,0]
    wmma_gemm<EPI_TANH_DW3><<<gFull, 256, 0, stream>>>(
        h1bf, DH, W2T, DH, DH, nullptr, 0, g2bf, DH, nullptr, 0,
        b2, W3, nullptr, 0, nullptr, 0, nullptr, 0.0f);
    // g1 = (1 - h1^2) * (g2 @ W2^T)
    wmma_gemm<EPI_OMH2><<<gFull, 256, 0, stream>>>(
        g2bf, DH, W2bf, DH, DH, nullptr, 0, g1bf, DH, nullptr, 0,
        nullptr, nullptr, h1bf, DH, nullptr, 0, nullptr, 0.0f);
    // grad slice + symplectic update (fused AXPY epilogue, dt from device)
    if (ev == 0) {        // dHdq at (q,p): p_half = p - 0.5*dt*grad ; -> z[:,512:1024]
      wmma_gemm<EPI_AXPY><<<gHalf, 256, 0, stream>>>(
          g1bf, DH, W1bf, DH, DH, p_half, SDIM, zbf + SDIM, ZDIM, nullptr, 0,
          nullptr, nullptr, nullptr, 0, pf, SDIM, s_dt, -0.5f);
    } else if (ev == 1) { // dHdp at (q,p_half): q_next = q + dt*grad ; -> z[:,0:512], qp[:,0:512]
      wmma_gemm<EPI_AXPY><<<gHalf, 256, 0, stream>>>(
          g1bf, DH, W1bf + (size_t)SDIM * DH, DH, DH, q_next, SDIM, zbf, ZDIM,
          qpbf, IDIM, nullptr, nullptr, nullptr, 0, qf, SDIM, s_dt, 1.0f);
    } else {              // dHdq at (q_next,p_half): p_next = p_half - 0.5*dt*grad ; -> qp[:,512:1024]
      wmma_gemm<EPI_AXPY><<<gHalf, 256, 0, stream>>>(
          g1bf, DH, W1bf, DH, DH, p_next, SDIM, qpbf + SDIM, IDIM, nullptr, 0,
          nullptr, nullptr, nullptr, 0, p_half, SDIM, s_dt, -0.5f);
    }
  }

  // ---- out = [q_next, p_next] @ W_out + b_out ----
  wmma_gemm<EPI_BIASF><<<gOut, 256, 0, stream>>>(
      qpbf, IDIM, WoT, IDIM, IDIM, out_main, IDIM, nullptr, 0, nullptr, 0,
      b_out, nullptr, nullptr, 0, nullptr, 0, nullptr, 0.0f);

  // ---- h_next pack ----
  k_pack<<<nBZ, 256, 0, stream>>>(q_next, p_next, out_hnxt, B);
}
